// RNN_71803263254839
// MI455X (gfx1250) — matrix operand order repaired and checked by reading, not compiled, not run on hardware
//
#include <hip/hip_runtime.h>
#include <hip/hip_bf16.h>
#include <math.h>

// Problem dims (fixed by reference)
#define B_  32
#define S_  1024
#define D_  1024
#define H_  1024

typedef __attribute__((ext_vector_type(16))) __bf16          v16bf;
typedef __attribute__((ext_vector_type(2)))  __bf16          v2bf;
typedef __attribute__((ext_vector_type(8)))  float           v8f;
typedef __attribute__((ext_vector_type(4)))  float           f32x4;
typedef __attribute__((ext_vector_type(8)))  unsigned short  us8;
typedef __attribute__((ext_vector_type(16))) unsigned short  us16;
typedef __attribute__((ext_vector_type(4)))  unsigned int    u32x4;
typedef __attribute__((ext_vector_type(8)))  unsigned int    u32x8;

__device__ __forceinline__ unsigned short f2bf(float f) {
  union { float f; unsigned u; } v; v.f = f;
  // round-to-nearest-even bf16
  return (unsigned short)((v.u + 0x7FFFu + ((v.u >> 16) & 1u)) >> 16);
}

// Pack two f32 into one dword of two bf16 (lo in [15:0], hi in [31:16]) with
// RNE rounding. Uses v_cvt_pk_bf16_f32 when the toolchain exposes it; the
// manual path stays on full 32-bit ALU ops (no half-register v_mov_b16 churn).
__device__ __forceinline__ unsigned pk2bf(float lo, float hi) {
#if __has_builtin(__builtin_amdgcn_cvt_pk_bf16_f32)
  v2bf p = __builtin_amdgcn_cvt_pk_bf16_f32(lo, hi);
  return __builtin_bit_cast(unsigned, p);
#else
  unsigned ul = __builtin_bit_cast(unsigned, lo);
  unsigned uh = __builtin_bit_cast(unsigned, hi);
  ul = (ul + 0x7FFFu + ((ul >> 16) & 1u)) >> 16;
  uh = (uh + 0x7FFFu + ((uh >> 16) & 1u)) & 0xFFFF0000u;
  return uh | ul;
#endif
}

__device__ __forceinline__ u32x4 cvt_pk4(f32x4 a, f32x4 b) {
  u32x4 p;
  p[0] = pk2bf(a.x, a.y);
  p[1] = pk2bf(a.z, a.w);
  p[2] = pk2bf(b.x, b.y);
  p[3] = pk2bf(b.z, b.w);
  return p;
}

__device__ __forceinline__ v16bf frag_from_pk(u32x4 lo, u32x4 hi) {
  u32x8 t = __builtin_shufflevector(lo, hi, 0, 1, 2, 3, 4, 5, 6, 7);
  return __builtin_bit_cast(v16bf, t);
}

__device__ __forceinline__ v16bf make_frag(us8 lo, us8 hi) {
  us16 t = __builtin_shufflevector(lo, hi, 0,1,2,3,4,5,6,7,8,9,10,11,12,13,14,15);
  return __builtin_bit_cast(v16bf, t);
}

__device__ __forceinline__ v8f wmma_bf16(v16bf a, v16bf b, v8f c) {
  // D = A(16x32 bf16) * B(32x16 bf16) + C(16x16 f32)
  return __builtin_amdgcn_wmma_f32_16x16x32_bf16(false, a, false, b, (short)0, c,
                                                 false, false);
}

// ---------------------------------------------------------------------------
// TDM: DMA one 128(rows) x 32(cols) f32 tile of X into LDS, with descriptor
// padding of 4 DWORDs after every 32 DWORDs -> LDS row stride 36 floats
// (bank-spread, 16B aligned). Issued by one wave; tracked by TENSORcnt.
// D# layout per CDNA5 ISA ch.8 (group0: count/lds_addr/global_addr/type,
// group1: data_size, pad, tensor dims/strides, tile dims).
// ---------------------------------------------------------------------------
__device__ __forceinline__ void tdm_load_Atile(const float* gsrc, unsigned lds_byte_addr) {
  unsigned long long ga = (unsigned long long)(size_t)gsrc;
  u32x4 g0;
  g0[0] = 1u;                                   // count=1, user mode, no gather
  g0[1] = lds_byte_addr;                        // lds_addr [63:32]
  g0[2] = (unsigned)ga;                         // global_addr [95:64]
  g0[3] = (unsigned)(ga >> 32) | 0x80000000u;   // global_addr[56:32] | type=2
  u32x8 g1;
  g1[0] = (2u << 16)          // data_size = 2 (4 bytes)
        | (1u << 20)          // pad_enable
        | (4u << 22)          // pad_interval code 4 = every 32 DWORDs
        | (3u << 25);         // pad_amount  code 3 = 4 DWORDs
  g1[1] = (unsigned)D_ << 16;         // tensor_dim0 = 1024 (bits 79:48, low16)
  g1[2] = (unsigned)(B_ * S_) << 16;  // tensor_dim1 = 32768 (bits 111:80, low16)
  g1[3] = 32u << 16;                  // tile_dim0 = 32 (bits 127:112)
  g1[4] = 128u;                       // tile_dim1 = 128, tile_dim2 = 0
  g1[5] = (unsigned)D_;               // tensor_dim0_stride = 1024 (low 32)
  g1[6] = 0u;                         // stride0 hi16 = 0, stride1 lo16 = 0
  g1[7] = 0u;                         // stride1 hi
  asm volatile("tensor_load_to_lds %0, %1" :: "s"(g0), "s"(g1) : "memory");
}

// ---------------------------------------------------------------------------
// Phase 0: one-time weight conversion.
//   WxT[n*K + k] = bf16(Wx[k*N + n])   (transposed: column-contiguous WMMA B)
//   WhT[n*K + k] = bf16(Wh[k*N + n])
//   hb0[b*H + h] = bf16(h0[h])
// ---------------------------------------------------------------------------
__global__ __launch_bounds__(256)
void rnn_convert(const float* __restrict__ Wx, const float* __restrict__ Wh,
                 const float* __restrict__ h0,
                 unsigned short* __restrict__ WxT,
                 unsigned short* __restrict__ WhT,
                 unsigned short* __restrict__ hb0) {
  int i = blockIdx.x * 256 + threadIdx.x;     // 0 .. 1M-1
  int n = i >> 10;
  int k = i & (D_ - 1);
  size_t src = (size_t)k * H_ + n;
  WxT[i] = f2bf(Wx[src]);
  WhT[i] = f2bf(Wh[src]);
  if (i < B_ * H_) hb0[i] = f2bf(h0[i & (H_ - 1)]);
}

// ---------------------------------------------------------------------------
// Phase 1: Pre = X @ Wx + bias (M=32768, N=1024, K=1024) -> d_out (f32).
// Block tile 128x128, BK=32. 8 waves as 2(m) x 4(n); wave = 4x2 WMMA tiles.
// A tile: TDM tensor_load_to_lds (f32, double-buffered, DMA overlaps compute);
// f32->bf16 pack happens at LDS->VGPR fragment load (dword-packed, RNE).
// B: direct bf16 loads from pre-transposed WxT (L2-resident) + prefetch.
// ---------------------------------------------------------------------------
#define BM 128
#define BN 128
#define BK 32
#define LDF 36          // padded LDS row stride in floats (from TDM pad fields)
#define NKS (D_ / BK)   // 32 k-steps

__global__ __launch_bounds__(256)
void rnn_xw_gemm(const float* __restrict__ X,
                 const unsigned short* __restrict__ WxT,
                 const float* __restrict__ bias,
                 float* __restrict__ out) {
  __shared__ float Af[2][BM][LDF];

  const int tid  = threadIdx.x;
  const int lane = tid & 31;
  const int wave = tid >> 5;
  const int wm   = wave & 1;    // 0..1 : 64-row slab
  const int wn   = wave >> 1;   // 0..3 : 32-col slab
  const int r    = lane & 15;
  const int kh   = lane >> 4;

  const int m0 = blockIdx.y * BM;
  const int n0 = blockIdx.x * BN;
  const float* Xblk = X + (size_t)m0 * D_;

  if (wave == 0)
    tdm_load_Atile(Xblk, (unsigned)(size_t)&Af[0][0][0]);

  v8f acc[4][2] = {};

  for (int ks = 0; ks < NKS; ++ks) {
    // Per-wave counter: immediate for waves 1..7, real wait only for the
    // issuing wave -> no EXEC save/restore divergence on the hot path.
    __builtin_amdgcn_s_wait_tensorcnt(0);
    __syncthreads();                                      // stage visible to all
    if (wave == 0 && ks + 1 < NKS)                        // overlap next DMA
      tdm_load_Atile(Xblk + (ks + 1) * BK,
                     (unsigned)(size_t)&Af[(ks + 1) & 1][0][0]);

    const float (*Ab)[LDF] = Af[ks & 1];
    const int kt = ks * BK;

    // A fragments (16x32): per lane K = kh*8..+7 and 16+kh*8..+7, f32->bf16
    v16bf a[4];
    #pragma unroll
    for (int fm = 0; fm < 4; ++fm) {
      const float* ar = &Ab[wm * 64 + fm * 16 + r][kh * 8];
      u32x4 lo = cvt_pk4(*(const f32x4*)ar,        *(const f32x4*)(ar + 4));
      u32x4 hi = cvt_pk4(*(const f32x4*)(ar + 16), *(const f32x4*)(ar + 20));
      a[fm] = frag_from_pk(lo, hi);
    }

    // B fragments (32x16 bf16): per lane-column, K = kh*8..+7 and 16+kh*8..+7
    #pragma unroll
    for (int fn = 0; fn < 2; ++fn) {
      int n = n0 + wn * 32 + fn * 16 + r;
      const unsigned short* bp = WxT + (size_t)n * D_ + kt + kh * 8;
      __builtin_prefetch(bp + BK, 0, 1);      // global_prefetch_b8, next k-step
      v16bf b = make_frag(*(const us8*)bp, *(const us8*)(bp + 16));
      #pragma unroll
      for (int fm = 0; fm < 4; ++fm)
        acc[fm][fn] = wmma_bf16(a[fm], b, acc[fm][fn]);
    }
    // no trailing barrier needed: next overwrite of this buffer happens only
    // after every wave passes the *next* iteration's top barrier
  }

  // Epilogue: add bias, store f32 pre-activations
  #pragma unroll
  for (int fn = 0; fn < 2; ++fn) {
    int n = n0 + wn * 32 + fn * 16 + r;
    float bv = bias[n];
    #pragma unroll
    for (int fm = 0; fm < 4; ++fm) {
      #pragma unroll
      for (int j = 0; j < 8; ++j) {
        int m = m0 + wm * 64 + fm * 16 + kh * 8 + j;
        out[(size_t)m * H_ + n] = acc[fm][fn][j] + bv;
      }
    }
  }
}

// ---------------------------------------------------------------------------
// Phase 2: one recurrent step:  h_t = tanh(Pre[:,t,:] + h_{t-1} @ Wh).
// Latency-bound -> split-K: 32 blocks x 8 waves; wave = (n-tile, k-chunk of 256).
// Dependent WMMA chain per accumulator is 8 (was 32); 4 K-partials reduced via
// LDS, then tanh + in-place d_out update + bf16 state for t+1 (double-buffered,
// so no cross-block races). Wh (2MB bf16) stays L2-resident across all steps.
// ---------------------------------------------------------------------------
__global__ __launch_bounds__(256)
void rnn_step(const unsigned short* __restrict__ hprev,   // [32][1024] bf16
              unsigned short* __restrict__ hnext,         // [32][1024] bf16
              const unsigned short* __restrict__ WhT,     // [N][K] bf16 transposed
              float* __restrict__ out, int t) {
  __shared__ float Ls[2][4][2][256];   // [ntile][kchunk][mtile][j*32+lane]

  const int tid  = threadIdx.x;
  const int lane = tid & 31;
  const int wave = tid >> 5;
  const int r    = lane & 15;
  const int kh   = lane >> 4;
  const int nt   = wave >> 2;                  // 0..1
  const int kc   = wave & 3;                   // 0..3 (k-chunk of 256)
  const int nAbs = blockIdx.x * 32 + nt * 16;

  v8f acc0 = {}, acc1 = {};
  const unsigned short* a0p = hprev + (size_t)r * H_;
  const unsigned short* a1p = hprev + (size_t)(16 + r) * H_;
  const unsigned short* bp  = WhT + (size_t)(nAbs + r) * H_;

  const int k0 = kc * 256;
  #pragma unroll 2
  for (int k = k0; k < k0 + 256; k += 32) {
    __builtin_prefetch(bp + k + 128, 0, 1);    // global_prefetch_b8 ahead on Wh
    int ao = k + kh * 8;
    v16bf a0 = make_frag(*(const us8*)(a0p + ao), *(const us8*)(a0p + ao + 16));
    v16bf a1 = make_frag(*(const us8*)(a1p + ao), *(const us8*)(a1p + ao + 16));
    int bo = k + kh * 8;
    v16bf b  = make_frag(*(const us8*)(bp + bo), *(const us8*)(bp + bo + 16));
    acc0 = wmma_bf16(a0, b, acc0);
    acc1 = wmma_bf16(a1, b, acc1);
  }

  #pragma unroll
  for (int j = 0; j < 8; ++j) {
    Ls[nt][kc][0][j * 32 + lane] = acc0[j];
    Ls[nt][kc][1][j * 32 + lane] = acc1[j];
  }
  __syncthreads();

  // Reduce 4 K-partials, fuse tanh epilogue, emit bf16 state for step t+1
  #pragma unroll
  for (int i = 0; i < 4; ++i) {
    int e   = tid + i * 256;          // 0..1023 = 32m x 32n block outputs
    int ent = e >> 9;
    int emt = (e >> 8) & 1;
    int pos = e & 255;
    int j   = pos >> 5;
    int l   = pos & 31;
    float s = Ls[ent][0][emt][pos] + Ls[ent][1][emt][pos]
            + Ls[ent][2][emt][pos] + Ls[ent][3][emt][pos];
    int m = ((l >> 4) * 8 + j) + emt * 16;
    int n = blockIdx.x * 32 + ent * 16 + (l & 15);
    size_t oi = ((size_t)m * S_ + t) * H_ + n;
    float hv = tanhf(out[oi] + s);
    out[oi] = hv;
    hnext[m * H_ + n] = f2bf(hv);
  }

  // Cluster-dispatch ready: under a clustered launch this is the HW cluster
  // barrier (s_barrier_signal/wait -3); under a plain dispatch ClusterID==0
  // and it is an architectural NOP.
  if (__builtin_amdgcn_cluster_id_x() > 0)
    __builtin_amdgcn_s_cluster_barrier();
}

// ---------------------------------------------------------------------------
// Launch: convert -> big GEMM -> 1024 sequential step kernels (graph-friendly).
// Workspace (bf16 as ushort): [0,1M) WxT | [1M,2M) WhT | 2x 32K h buffers.
// ---------------------------------------------------------------------------
extern "C" void kernel_launch(void* const* d_in, const int* in_sizes, int n_in,
                              void* d_out, int out_size, void* d_ws, size_t ws_size,
                              hipStream_t stream) {
  const float* x   = (const float*)d_in[0];   // [B,S,D]
  const float* Wx  = (const float*)d_in[1];   // [D,H]
  const float* Wh  = (const float*)d_in[2];   // [H,H]
  const float* bia = (const float*)d_in[3];   // [H]
  const float* h0  = (const float*)d_in[4];   // [H]
  float* out = (float*)d_out;                 // [B,S,H]

  unsigned short* WxT = (unsigned short*)d_ws;
  unsigned short* WhT = WxT + (size_t)D_ * H_;
  unsigned short* hb0 = WhT + (size_t)H_ * H_;
  unsigned short* hb1 = hb0 + (size_t)B_ * H_;

  (void)in_sizes; (void)n_in; (void)out_size; (void)ws_size;

  rnn_convert<<<(D_ * H_) / 256, 256, 0, stream>>>(Wx, Wh, h0, WxT, WhT, hb0);

  dim3 g1(H_ / BN, (B_ * S_) / BM);
  rnn_xw_gemm<<<g1, 256, 0, stream>>>(x, WxT, bia, out);

  for (int t = 0; t < S_; ++t) {
    const unsigned short* hp = (t & 1) ? hb1 : hb0;
    unsigned short*       hn = (t & 1) ? hb0 : hb1;
    rnn_step<<<H_ / 32, 256, 0, stream>>>(hp, hn, WhT, out, t);
  }
}
